// CategoricalCNN_56659208569217
// MI455X (gfx1250) — matrix operand order repaired and checked by reading, not compiled, not run on hardware
//
#include <hip/hip_runtime.h>
#include <math.h>

typedef __attribute__((ext_vector_type(16))) _Float16 v16h;
typedef __attribute__((ext_vector_type(8)))  _Float16 v8h;
typedef __attribute__((ext_vector_type(8)))  float    v8f;

#define C_IN   3
#define H_IN   1280
#define W_IN   720
#define BS     20
#define SF     4
#define NH     64
#define NW     36
#define NBLK   (NH * NW)          // 2304
#define OUTC   48                 // light expert conv channels
#define H_OUT  (H_IN * SF)        // 5120
#define W_OUT  (W_IN * SF)        // 2880
#define IMG_OUT_ELEMS ((size_t)C_IN * H_OUT * W_OUT)   // 44,236,800

// im2col constant offset for 3x3 conv, kernel index k, padded row stride S.
#define IM2COL_CST(k, S, PLANE) (((k) / 9) * (PLANE) + (((k) % 9) / 3) * (S) + ((k) % 3))

// ---------------------------------------------------------------------------
// Gate CNN stage 1 as WMMA implicit GEMM:
//   conv 3->16 (3x3, replicate pad) == C[16 x 921600] = W[16 x 27pad32] * im2col
// 16x16 conv tile per workgroup (grid 80x45), so position n = (y=nt, x=lane&15)
// needs no div/mod. B-matrix built once in LDS in fragment-native [n][k]
// layout (constant DS offsets), fragments loaded as 32B LDS vector loads.
// Conv f32 results -> LDS, fused maxpool2 + tanh (tanh monotone).
// Output: t1 [16][640][360]
// ---------------------------------------------------------------------------
__global__ void __launch_bounds__(128) gate1_kernel(
        const float* __restrict__ in,
        const float* __restrict__ w1,
        const float* __restrict__ b1,
        float* __restrict__ t1) {
    __shared__ __align__(16) _Float16 smW[16 * 32];    // weights K-pad 27->32
    __shared__ float    smB[16];
    __shared__ _Float16 smIn[3 * 18 * 18];             // replicate-padded tile
    __shared__ __align__(32) _Float16 smBm[256 * 32];  // im2col B (16 KB)
    __shared__ float    smC[16 * 256];                 // conv results (16 KB)

    const int tid = threadIdx.x;
    const int ty  = blockIdx.x / 45;
    const int tx  = blockIdx.x % 45;
    const int y0  = ty * 16;
    const int x0  = tx * 16;

    for (int i = tid; i < 16 * 32; i += 128) {
        int row = i >> 5, k = i & 31;
        smW[i] = (k < 27) ? (_Float16)w1[row * 27 + k] : (_Float16)0.0f;
    }
    if (tid < 16) smB[tid] = b1[tid];
    for (int i = tid; i < 3 * 18 * 18; i += 128) {
        int c = i / 324, r = i % 324, py = r / 18, px = r % 18;
        int iy = y0 + py - 1; iy = iy < 0 ? 0 : (iy > H_IN - 1 ? H_IN - 1 : iy);
        int ix = x0 + px - 1; ix = ix < 0 ? 0 : (ix > W_IN - 1 ? W_IN - 1 : ix);
        smIn[i] = (_Float16)in[((size_t)c * H_IN + iy) * W_IN + ix];
    }
    __syncthreads();

    // Build B matrix: position n, k contiguous; all DS offsets constant.
    for (int n = tid; n < 256; n += 128) {
        int base = (n >> 4) * 18 + (n & 15);
        v16h blo, bhi;
        #pragma unroll
        for (int e = 0; e < 16; ++e)
            blo[e] = smIn[IM2COL_CST(e, 18, 324) + base];
        #pragma unroll
        for (int e = 0; e < 16; ++e)
            bhi[e] = (16 + e < 27)
                   ? smIn[IM2COL_CST(16 + e, 18, 324) + base]
                   : (_Float16)0.0f;
        *(v16h*)&smBm[n * 32]      = blo;
        *(v16h*)&smBm[n * 32 + 16] = bhi;
    }
    __syncthreads();

    const int  lane = tid & 31;       // wave32
    const int  wave = tid >> 5;
    const int  ln16 = lane & 15;
    const bool lo   = lane < 16;
    const int  asel = lo ? 0 : 16;    // A: K 0..15 vs 16..31
    const int  kb   = lo ? 0 : 16;    // B: K 0..15 vs 16..31

    v8h a0 = *(const v8h*)&smW[ln16 * 32 + asel];
    v8h a1 = *(const v8h*)&smW[ln16 * 32 + 8 + asel];
    v16h afrag = __builtin_shufflevector(a0, a1, 0, 1, 2, 3, 4, 5, 6, 7,
                                         8, 9, 10, 11, 12, 13, 14, 15);
    float biasr[8];
    #pragma unroll
    for (int v = 0; v < 8; ++v) biasr[v] = smB[lo ? v : v + 8];

    for (int nt = wave; nt < 16; nt += 4) {           // 256 pos = 16 n-tiles
        int n = nt * 16 + ln16;                        // y = nt, x = ln16
        v16h bfrag = *(const v16h*)&smBm[n * 32 + kb];
        v8f acc;
        #pragma unroll
        for (int v = 0; v < 8; ++v) acc[v] = biasr[v];
        acc = __builtin_amdgcn_wmma_f32_16x16x32_f16(
                  false, afrag, false, bfrag, (short)0, acc, false, false);
        #pragma unroll
        for (int v = 0; v < 8; ++v)
            smC[(lo ? v : v + 8) * 256 + n] = acc[v];
    }
    __syncthreads();

    // Fused maxpool2 + tanh: pooled tile 8x8 x 16 channels = 1024 outputs.
    for (int i = tid; i < 16 * 64; i += 128) {
        int oc = i >> 6, r = i & 63, oy = r >> 3, ox = r & 7;
        const float* p = &smC[oc * 256 + (2 * oy) * 16 + 2 * ox];
        float m = fmaxf(fmaxf(p[0], p[1]), fmaxf(p[16], p[17]));
        int gy = ty * 8 + oy, gx = tx * 8 + ox;
        t1[((size_t)oc * 640 + gy) * 360 + gx] = tanhf(m);
    }
}

// ---------------------------------------------------------------------------
// Gate CNN stage 2: conv 16->8 (3x3, replicate pad) + maxpool2.
// Channel-batched: one thread per pooled pixel computes all 8 out-channels
// from one clamped 4x4 window per in-channel (weights staged in LDS).
// t1 [16][640][360] -> t2 [8][320][180]
// ---------------------------------------------------------------------------
__global__ void __launch_bounds__(256) gate2_kernel(
        const float* __restrict__ t1,
        const float* __restrict__ w2,
        const float* __restrict__ b2,
        float* __restrict__ t2) {
    const int HI = 640, WI = 360, HO = 320, WO = 180;
    __shared__ float smW[8 * 16 * 9];   // 1152 floats
    const int tid = threadIdx.x;
    for (int i = tid; i < 8 * 16 * 9; i += 256) smW[i] = w2[i];
    __syncthreads();

    int idx = blockIdx.x * 256 + tid;
    if (idx >= HO * WO) return;
    int x = idx % WO, y = idx / WO;

    int ry[4], rx[4];
    #pragma unroll
    for (int r = 0; r < 4; ++r) {
        int iy = 2 * y + r - 1; ry[r] = iy < 0 ? 0 : (iy > HI - 1 ? HI - 1 : iy);
        int ix = 2 * x + r - 1; rx[r] = ix < 0 ? 0 : (ix > WI - 1 ? WI - 1 : ix);
    }

    float acc[8][4];
    #pragma unroll
    for (int oc = 0; oc < 8; ++oc) {
        float b = b2[oc];
        #pragma unroll
        for (int p = 0; p < 4; ++p) acc[oc][p] = b;
    }

    for (int c = 0; c < 16; ++c) {
        float win[4][4];
        #pragma unroll
        for (int r = 0; r < 4; ++r)
        #pragma unroll
        for (int q = 0; q < 4; ++q)
            win[r][q] = t1[((size_t)c * HI + ry[r]) * WI + rx[q]];
        #pragma unroll
        for (int oc = 0; oc < 8; ++oc) {
            const float* wp = &smW[(oc * 16 + c) * 9];
            #pragma unroll
            for (int i = 0; i < 2; ++i)
            #pragma unroll
            for (int j = 0; j < 2; ++j) {
                float a = acc[oc][i * 2 + j];
                #pragma unroll
                for (int ky = 0; ky < 3; ++ky)
                #pragma unroll
                for (int kx = 0; kx < 3; ++kx)
                    a = fmaf(win[i + ky][j + kx], wp[ky * 3 + kx], a);
                acc[oc][i * 2 + j] = a;
            }
        }
    }
    #pragma unroll
    for (int oc = 0; oc < 8; ++oc) {
        float m = fmaxf(fmaxf(acc[oc][0], acc[oc][1]),
                        fmaxf(acc[oc][2], acc[oc][3]));
        t2[((size_t)oc * HO + y) * WO + x] = m;
    }
}

// ---------------------------------------------------------------------------
// Gate CNN stage 3: conv 8->1 (5x5, stride 5, VALID) + sigmoid -> class_vector
// ---------------------------------------------------------------------------
__global__ void gate3_kernel(const float* __restrict__ t2,
                             const float* __restrict__ w3,
                             const float* __restrict__ b3,
                             float* __restrict__ cls) {
    const int HI = 320, WI = 180;
    int idx = blockIdx.x * blockDim.x + threadIdx.x;
    if (idx >= NBLK) return;
    int bx = idx % NW;
    int by = idx / NW;
    float acc = b3[0];
    for (int c = 0; c < 8; ++c) {
        #pragma unroll
        for (int ky = 0; ky < 5; ++ky)
        #pragma unroll
        for (int kx = 0; kx < 5; ++kx) {
            acc = fmaf(t2[((size_t)c * HI + by * 5 + ky) * WI + bx * 5 + kx],
                       w3[(c * 5 + ky) * 5 + kx], acc);
        }
    }
    cls[idx] = 1.0f / (1.0f + expf(-acc));
}

// ---------------------------------------------------------------------------
// Light expert via WMMA implicit GEMM.
// Per 20x20 block:  C[48 x 400] = W[48 x 27pad32] * im2col[27pad32 x 400]
//   -> 3 M-tiles x 25 N-tiles of v_wmma_f32_16x16x32_f16, single K step.
// Pipeline: raw block via GLOBAL_LOAD_ASYNC_TO_LDS_B32 (ASYNCcnt path) ->
// zero-padded f16 image -> fragment-native B matrix smBm[n][k] (constant DS
// offsets; 32B vector stores) -> per-tile fragment = one 32B LDS load.
// Pixel-shuffle(4) + clamp[0,0.6]+0.4 fused into C write-out; NT b128 stores
// (177 MB streamed output, never re-read -> don't pollute L2).
// ---------------------------------------------------------------------------
__global__ void __launch_bounds__(128) light_expert_kernel(
        const float* __restrict__ in,
        const float* __restrict__ wl,
        const float* __restrict__ bl,
        float* __restrict__ out) {
    __shared__ float    smRaw[3 * BS * BS];            // async DMA destination
    __shared__ __align__(16) _Float16 smW[OUTC * 32];  // weights K-pad 27->32
    __shared__ _Float16 smI[3 * 22 * 22];              // zero-padded f16 block
    __shared__ __align__(32) _Float16 smBm[400 * 32];  // im2col B (25.6 KB)
    __shared__ float    smB[OUTC];

    const int tid = threadIdx.x;
    const int blk = blockIdx.x;
    const int bh  = blk / NW;
    const int bw  = blk % NW;

    // Async global->LDS staging of the raw 3x20x20 block (CDNA5 ASYNC path).
    for (int i = tid; i < 3 * BS * BS; i += 128) {
        int c = i / 400, r = i % 400, y = r / 20, x = r % 20;
        const float* g =
            &in[((size_t)c * H_IN + bh * BS + y) * W_IN + bw * BS + x];
        unsigned int lds = (unsigned int)(uintptr_t)&smRaw[i];
        asm volatile("global_load_async_to_lds_b32 %0, %1, off"
                     :: "v"(lds), "v"(g) : "memory");
    }
    // Meanwhile stage f16 weights (48 x 32) + bias through the sync path.
    for (int i = tid; i < OUTC * 32; i += 128) {
        int row = i >> 5, k = i & 31;
        smW[i] = (k < 27) ? (_Float16)wl[row * 27 + k] : (_Float16)0.0f;
    }
    if (tid < OUTC) smB[tid] = bl[tid];

    asm volatile("s_wait_asynccnt 0x0" ::: "memory");
    __syncthreads();

    // Raw block -> zero-padded 3x22x22 f16 image.
    for (int i = tid; i < 3 * 484; i += 128) {
        int c = i / 484, r = i % 484, py = r / 22, px = r % 22;
        float v = 0.0f;
        if (py >= 1 && py <= 20 && px >= 1 && px <= 20)
            v = smRaw[c * 400 + (py - 1) * 20 + (px - 1)];
        smI[i] = (_Float16)v;
    }
    __syncthreads();

    // Build fragment-native B matrix: smBm[n][k], constant DS offsets.
    for (int n = tid; n < 400; n += 128) {
        int base = (n / 20) * 22 + (n % 20);
        v16h blo, bhi;
        #pragma unroll
        for (int e = 0; e < 16; ++e)
            blo[e] = smI[IM2COL_CST(e, 22, 484) + base];
        #pragma unroll
        for (int e = 0; e < 16; ++e)
            bhi[e] = (16 + e < 27)
                   ? smI[IM2COL_CST(16 + e, 22, 484) + base]
                   : (_Float16)0.0f;
        *(v16h*)&smBm[n * 32]      = blo;
        *(v16h*)&smBm[n * 32 + 16] = bhi;
    }
    __syncthreads();

    const int  lane = tid & 31;      // wave32
    const int  wave = tid >> 5;
    const int  ln16 = lane & 15;
    const bool lo   = lane < 16;
    const int  asel = lo ? 0 : 16;   // A: K 0..15 vs 16..31
    const int  kb   = lo ? 0 : 16;   // B: K 0..15 vs 16..31

    // A fragments (3 m-tiles) via wide LDS vector loads.
    v16h afrag[3];
    float biasr[3][8];
    #pragma unroll
    for (int mt = 0; mt < 3; ++mt) {
        int row = mt * 16 + ln16;
        v8h a0 = *(const v8h*)&smW[row * 32 + asel];
        v8h a1 = *(const v8h*)&smW[row * 32 + 8 + asel];
        afrag[mt] = __builtin_shufflevector(a0, a1, 0, 1, 2, 3, 4, 5, 6, 7,
                                            8, 9, 10, 11, 12, 13, 14, 15);
        #pragma unroll
        for (int v = 0; v < 8; ++v)
            biasr[mt][v] = smB[mt * 16 + (lo ? v : v + 8)];
    }

    for (int nt = wave; nt < 25; nt += 4) {
        int n = nt * 16 + ln16;              // spatial index, column of B
        int y = n / 20, x = n % 20;
        v16h bfrag = *(const v16h*)&smBm[n * 32 + kb];
        #pragma unroll
        for (int mt = 0; mt < 3; ++mt) {
            v8f acc;
            #pragma unroll
            for (int v = 0; v < 8; ++v) acc[v] = biasr[mt][v];
            acc = __builtin_amdgcn_wmma_f32_16x16x32_f16(
                      false, afrag[mt], false, bfrag,
                      (short)0, acc, false, false);
            // Write-out: channel = mt, sub = M%16 = r1*4+r2 (pixel shuffle).
            #pragma unroll
            for (int v = 0; v < 8; ++v) {
                int   sub = lo ? v : v + 8;
                int   r1  = sub >> 2, r2 = sub & 3;
                float res = fminf(fmaxf(acc[v], 0.0f), 0.6f) + 0.4f;
                int   gy  = bh * 80 + y * 4 + r1;
                int   gx  = bw * 80 + x * 4 + r2;
                __builtin_nontemporal_store(
                    res, &out[((size_t)mt * H_OUT + gy) * W_OUT + gx]);
            }
        }
    }
}

// ---------------------------------------------------------------------------
extern "C" void kernel_launch(void* const* d_in, const int* in_sizes, int n_in,
                              void* d_out, int out_size, void* d_ws, size_t ws_size,
                              hipStream_t stream) {
    const float* in = (const float*)d_in[0];
    const float* w1 = (const float*)d_in[1];
    const float* b1 = (const float*)d_in[2];
    const float* w2 = (const float*)d_in[3];
    const float* b2 = (const float*)d_in[4];
    const float* w3 = (const float*)d_in[5];
    const float* b3 = (const float*)d_in[6];
    const float* wl = (const float*)d_in[7];
    const float* bl = (const float*)d_in[8];
    // d_in[9..14] = complex-expert weights: dead code (mask is always False).

    float* out = (float*)d_out;
    float* t1  = (float*)d_ws;                     // [16][640][360] = 14.75 MB
    float* t2  = t1 + (size_t)16 * 640 * 360;      // [8][320][180]  =  1.84 MB

    // Output image (bandwidth-bound part, WMMA + async-LDS + NT stores).
    light_expert_kernel<<<NBLK, 128, 0, stream>>>(in, wl, bl, out);

    // Gate CNN -> class_vector at tail of d_out.
    gate1_kernel<<<80 * 45, 128, 0, stream>>>(in, w1, b1, t1);
    gate2_kernel<<<(320 * 180 + 255) / 256, 256, 0, stream>>>(t1, w2, b2, t2);
    gate3_kernel<<<(NBLK + 255) / 256, 256, 0, stream>>>(t2, w3, b3,
                                                         out + IMG_OUT_ELEMS);
}
